// DeepPointNetKAN_3607772529059
// MI455X (gfx1250) — matrix operand order repaired and checked by reading, not compiled, not run on hardware
//
#include <hip/hip_runtime.h>
#include <hip/hip_bf16.h>
#include <cstdint>
#include <cstddef>

// ---------------------------------------------------------------------------
// Types / constants
// ---------------------------------------------------------------------------
typedef __bf16 bf16;
typedef __attribute__((ext_vector_type(16))) __bf16 v16bf;
typedef __attribute__((ext_vector_type(8)))  float  v8f;

#define BSZ   8
#define NPTS  1024
#define MROWS 8192          // B * N
#define NUMOBJ 16

#define BM  128
#define BNN 128
#define BK  32
#define LDSROW (BK + 8)                 // 40 elems -> 80B row stride (16B mult)
#define SBUF_E ((BM + BNN) * LDSROW)    // elements per double-buffer stage
#define B_OFF_E (BM * LDSROW)

static inline int cdiv(int a, int b) { return (a + b - 1) / b; }
static inline int KP(int I) { int k = 5 * I; return (k + 31) & ~31; }   // K padded to BK
static inline int OPD(int O) { return (O + 127) & ~127; }               // O padded to BNN

// ---------------------------------------------------------------------------
// CDNA5 async global->LDS copy (ASYNCcnt-tracked), per-lane 16B
// lds_off: byte offset within LDS (generic shared addr truncated to 32b)
// ---------------------------------------------------------------------------
__device__ __forceinline__ void async_copy16(uint32_t lds_off, uint64_t gaddr) {
  asm volatile("global_load_async_to_lds_b128 %0, %1, off"
               :: "v"(lds_off), "v"(gaddr) : "memory");
}
__device__ __forceinline__ void wait_async4() {
  asm volatile("s_wait_asynccnt 0x4" ::: "memory");
}
__device__ __forceinline__ void wait_async0() {
  asm volatile("s_wait_asynccnt 0x0" ::: "memory");
}

// ---------------------------------------------------------------------------
// Jacobi basis, a=b=1, degree 4 (matches reference recurrence)
// ---------------------------------------------------------------------------
__device__ __forceinline__ void jacobi5(float t, float P[5]) {
  P[0] = 1.0f;
  P[1] = 2.0f * t;   // (a-b + (a+b+2)x)/2 with a=b=1
#pragma unroll
  for (int i = 2; i <= 4; ++i) {
    float fi = (float)i;
    float d  = 2.0f * fi * (fi + 2.0f);                       // 2i(i+a+b)
    float A  = (2.0f * fi + 1.0f) * (2.0f * fi + 2.0f) / d;   // (2i+1)(2i+2)/d
    float Cc = -2.0f * fi * fi * (2.0f * fi + 2.0f) / (d * 2.0f * fi);
    P[i] = A * t * P[i - 1] + Cc * P[i - 2];
  }
}

// ---------------------------------------------------------------------------
// Weight conversion: W fp32 (I,O,5) -> Wb bf16 TRANSPOSED [Opad, Kpad]
// (B^T layout so GEMM B-tiles stage exactly like A-tiles and fragment
//  loads are per-lane K-contiguous -> ds_load_b128)
// ---------------------------------------------------------------------------
__global__ void wconv(const float* __restrict__ W, bf16* __restrict__ Wb,
                      int I, int O, int Kpad, int Opad) {
  int idx = blockIdx.x * 256 + threadIdx.x;
  if (idx >= Opad * Kpad) return;
  int o = idx / Kpad, k = idx % Kpad;
  float v = 0.0f;
  if (o < O && k < I * 5) {
    int i = k / 5, d = k % 5;
    v = W[((size_t)i * O + o) * 5 + d];
  }
  Wb[idx] = (bf16)v;
}

// ---------------------------------------------------------------------------
// Jacobi feature expansion: activation -> bf16 A matrix [MROWS, Kpad]
// ---------------------------------------------------------------------------
__global__ void expand_mc(const float* __restrict__ Y, bf16* __restrict__ Aout,
                          int C, int Kpad) {
  int idx = blockIdx.x * 256 + threadIdx.x;
  if (idx >= MROWS * C) return;
  int m = idx / C, i = idx % C;
  float t = tanhf(Y[(size_t)m * C + i]);
  float P[5]; jacobi5(t, P);
  size_t base = (size_t)m * Kpad + (size_t)i * 5;
#pragma unroll
  for (int d = 0; d < 5; ++d) Aout[base + d] = (bf16)P[d];
}

__global__ void expand_bcn(const float* __restrict__ X, bf16* __restrict__ Aout,
                           int C, int Kpad) {
  int idx = blockIdx.x * 256 + threadIdx.x;
  if (idx >= MROWS * C) return;
  int m = idx / C, i = idx % C;
  int b = m / NPTS, n = m % NPTS;
  float t = tanhf(X[((size_t)b * C + i) * NPTS + n]);
  float P[5]; jacobi5(t, P);
  size_t base = (size_t)m * Kpad + (size_t)i * 5;
#pragma unroll
  for (int d = 0; d < 5; ++d) Aout[base + d] = (bf16)P[d];
}

__global__ void padk(bf16* __restrict__ Aout, int k5, int Kpad) {
  int pad = Kpad - k5;
  int idx = blockIdx.x * 256 + threadIdx.x;
  if (idx >= MROWS * pad) return;
  int m = idx / pad, j = idx % pad;
  Aout[(size_t)m * Kpad + k5 + j] = (bf16)0.0f;
}

// ---------------------------------------------------------------------------
// bf16 WMMA GEMM: C[M,O] = A[M,Kpad] x B^T[Opad,Kpad]^T
// block tile 128x128x32, 8 wave32 waves (2x4), wave tile 64x32 (4x2 frags)
// double-buffered LDS fed by global_load_async_to_lds_b128 (ASYNCcnt);
// all fragments preloaded so the 8 WMMAs issue back-to-back per K-tile
// ---------------------------------------------------------------------------
__global__ __launch_bounds__(256) void gemm_bf16_wmma(
    const bf16* __restrict__ A, const bf16* __restrict__ Bw,
    float* __restrict__ C, int Kpad, int O, int Opad) {
  __shared__ __align__(16) bf16 smem[2 * SBUF_E];   // 40KB: A|B double buffered

  const int tid  = threadIdx.x;
  const int lane = tid & 31;
  const int wave = tid >> 5;
  const int wm   = wave >> 2;            // 0..1  -> 64 rows each
  const int wn   = wave & 3;             // 0..3  -> 32 cols each
  const size_t m0 = (size_t)blockIdx.y * BM;
  const size_t n0 = (size_t)blockIdx.x * BNN;

  // staging map: thread -> rows r0 / r0+64, 8-elem chunk c8 (16B aligned)
  const int r0 = tid >> 2;
  const int c8 = (tid & 3) << 3;
  const uint32_t ldsbase = (uint32_t)(uintptr_t)smem;  // LDS offset = addr[31:0]

  uint32_t lA0[2], lA1[2], lB0[2], lB1[2];
#pragma unroll
  for (int s = 0; s < 2; ++s) {
    lA0[s] = ldsbase + 2u * (s * SBUF_E + r0 * LDSROW + c8);
    lA1[s] = ldsbase + 2u * (s * SBUF_E + (r0 + 64) * LDSROW + c8);
    lB0[s] = ldsbase + 2u * (s * SBUF_E + B_OFF_E + r0 * LDSROW + c8);
    lB1[s] = ldsbase + 2u * (s * SBUF_E + B_OFF_E + (r0 + 64) * LDSROW + c8);
  }
  uint64_t gA0 = (uint64_t)(uintptr_t)(A  + (m0 + r0)      * (size_t)Kpad + c8);
  uint64_t gA1 = (uint64_t)(uintptr_t)(A  + (m0 + r0 + 64) * (size_t)Kpad + c8);
  uint64_t gB0 = (uint64_t)(uintptr_t)(Bw + (n0 + r0)      * (size_t)Kpad + c8);
  uint64_t gB1 = (uint64_t)(uintptr_t)(Bw + (n0 + r0 + 64) * (size_t)Kpad + c8);

  // prologue: stage tile 0 into buffer 0 (4 async ops per thread)
  async_copy16(lA0[0], gA0); async_copy16(lA1[0], gA1);
  async_copy16(lB0[0], gB0); async_copy16(lB1[0], gB1);
  gA0 += BK * 2; gA1 += BK * 2; gB0 += BK * 2; gB1 += BK * 2;

  v8f zero;
#pragma unroll
  for (int e = 0; e < 8; ++e) zero[e] = 0.0f;
  v8f acc[4][2];
#pragma unroll
  for (int i = 0; i < 4; ++i)
#pragma unroll
    for (int j = 0; j < 2; ++j) acc[i][j] = zero;

  const int ntiles = Kpad / BK;
  for (int kt = 0; kt < ntiles; ++kt) {
    const int cur = kt & 1;
    if (kt + 1 < ntiles) {
      const int nxt = cur ^ 1;
      async_copy16(lA0[nxt], gA0); async_copy16(lA1[nxt], gA1);
      async_copy16(lB0[nxt], gB0); async_copy16(lB1[nxt], gB1);
      gA0 += BK * 2; gA1 += BK * 2; gB0 += BK * 2; gB1 += BK * 2;
      wait_async4();                 // tile kt landed (4 newest = tile kt+1)
    } else {
      wait_async0();                 // last tile landed
    }
    __syncthreads();

    const bf16* Ab = &smem[cur * SBUF_E];
    const bf16* Bb = &smem[cur * SBUF_E + B_OFF_E];

    // Preload ALL fragments, then fire the 8 WMMAs back-to-back.
    // B fragments: lane col = lane&15; lanes 0-15 K{0..7,16..23}; lanes 16-31 K{8..15,24..31}
    v16bf bfrag[2];
#pragma unroll
    for (int nf = 0; nf < 2; ++nf) {
      const bf16* src = &Bb[(wn * 32 + nf * 16 + (lane & 15)) * LDSROW
                            + ((lane >> 4) << 3)];
#pragma unroll
      for (int j = 0; j < 16; ++j) bfrag[nf][j] = src[j < 8 ? j : 8 + j];
    }
    // A fragments: lanes 0-15 K{0..7,16..23}; lanes 16-31 K{8..15,24..31}
    v16bf afrag[4];
    const int koff = (lane >> 4) << 3;
#pragma unroll
    for (int mf = 0; mf < 4; ++mf) {
      const bf16* src = &Ab[(wm * 64 + mf * 16 + (lane & 15)) * LDSROW];
#pragma unroll
      for (int j = 0; j < 8; ++j) {
        afrag[mf][j]     = src[koff + j];
        afrag[mf][8 + j] = src[16 + koff + j];
      }
    }
#pragma unroll
    for (int mf = 0; mf < 4; ++mf)
#pragma unroll
      for (int nf = 0; nf < 2; ++nf)
        acc[mf][nf] = __builtin_amdgcn_wmma_f32_16x16x32_bf16(
            false, afrag[mf], false, bfrag[nf], (short)0, acc[mf][nf], false, false);
    __syncthreads();
  }

  // C/D layout: VGPR r -> row (lane<16 ? r : 8+r), col = lane&15
  const int nl    = lane & 15;
  const int mhalf = (lane >> 4) << 3;
#pragma unroll
  for (int mf = 0; mf < 4; ++mf)
#pragma unroll
    for (int nf = 0; nf < 2; ++nf) {
      int n = (int)n0 + wn * 32 + nf * 16 + nl;
      if (n < O) {
        size_t mb = m0 + wm * 64 + mf * 16 + mhalf;
#pragma unroll
        for (int r = 0; r < 8; ++r) C[(mb + r) * (size_t)O + n] = acc[mf][nf][r];
      }
    }
}

// ---------------------------------------------------------------------------
// BatchNorm (training mode, biased variance) over M=8192 rows, two-stage
// ---------------------------------------------------------------------------
__global__ void bn_stats1(const float* __restrict__ Y, int O,
                          float* __restrict__ psum, float* __restrict__ psq) {
  int idx = blockIdx.x * 256 + threadIdx.x;
  if (idx >= O * 32) return;
  int s = idx / O, c = idx % O;
  const int chunk = MROWS / 32;
  float su = 0.0f, sq = 0.0f;
  for (int m = s * chunk; m < (s + 1) * chunk; ++m) {
    float v = Y[(size_t)m * O + c];
    su += v; sq += v * v;
  }
  psum[idx] = su; psq[idx] = sq;
}

__global__ void bn_stats2(const float* __restrict__ psum, const float* __restrict__ psq,
                          int O, float* __restrict__ meanv, float* __restrict__ rsigv) {
  int c = blockIdx.x * 256 + threadIdx.x;
  if (c >= O) return;
  float su = 0.0f, sq = 0.0f;
  for (int k = 0; k < 32; ++k) { su += psum[k * O + c]; sq += psq[k * O + c]; }
  float mu  = su / (float)MROWS;
  float var = sq / (float)MROWS - mu * mu;
  meanv[c] = mu;
  rsigv[c] = rsqrtf(var + 1e-5f);
}

__global__ void bn_apply(float* __restrict__ Y, int O,
                         const float* __restrict__ meanv, const float* __restrict__ rsigv,
                         const float* __restrict__ g, const float* __restrict__ b) {
  int idx = blockIdx.x * 256 + threadIdx.x;
  if (idx >= MROWS * O) return;
  int c = idx % O;
  Y[idx] = (Y[idx] - meanv[c]) * rsigv[c] * g[c] + b[c];
}

// BN over batch dim only (kan outputs, 8 rows)
__global__ void bn_batch(float* __restrict__ Y, int O,
                         const float* __restrict__ g, const float* __restrict__ b) {
  int c = blockIdx.x * 256 + threadIdx.x;
  if (c >= O) return;
  float su = 0.0f, sq = 0.0f;
  for (int bb = 0; bb < BSZ; ++bb) { float v = Y[bb * O + c]; su += v; sq += v * v; }
  float mu  = su / (float)BSZ;
  float var = sq / (float)BSZ - mu * mu;
  float rs  = rsqrtf(var + 1e-5f);
  for (int bb = 0; bb < BSZ; ++bb)
    Y[bb * O + c] = (Y[bb * O + c] - mu) * rs * g[c] + b[c];
}

// ---------------------------------------------------------------------------
// Max pool over points: [M,O] -> [B,O]
// ---------------------------------------------------------------------------
__global__ void maxpool(const float* __restrict__ Y, int O, float* __restrict__ feat) {
  int idx = blockIdx.x * 256 + threadIdx.x;
  if (idx >= BSZ * O) return;
  int b = idx / O, c = idx % O;
  float mx = -3.4e38f;
  for (int n = 0; n < NPTS; ++n)
    mx = fmaxf(mx, Y[((size_t)b * NPTS + n) * O + c]);
  feat[idx] = mx;
}

// ---------------------------------------------------------------------------
// kan (B=8 rows): jacobi expand then fp32 GEMV
// ---------------------------------------------------------------------------
__global__ void kan_expand(const float* __restrict__ in, float* __restrict__ jbuf, int I) {
  int idx = blockIdx.x * 256 + threadIdx.x;
  if (idx >= BSZ * I) return;
  float t = tanhf(in[idx]);
  float P[5]; jacobi5(t, P);
#pragma unroll
  for (int d = 0; d < 5; ++d) jbuf[(size_t)idx * 5 + d] = P[d];
}

__global__ void kan_gemv(const float* __restrict__ jbuf, const float* __restrict__ W,
                         float* __restrict__ out, int I, int O) {
  int idx = blockIdx.x * 256 + threadIdx.x;
  if (idx >= BSZ * O) return;
  int b = idx / O, o = idx % O;
  float s = 0.0f;
  for (int i = 0; i < I; ++i) {
    const float* j5 = &jbuf[((size_t)b * I + i) * 5];
    const float* w5 = &W[((size_t)i * O + o) * 5];
#pragma unroll
    for (int d = 0; d < 5; ++d) s += j5[d] * w5[d];
  }
  out[idx] = s;
}

// ---------------------------------------------------------------------------
// T-net applications
// ---------------------------------------------------------------------------
__global__ void tnet3_apply(const float* __restrict__ X, const float* __restrict__ T,
                            float* __restrict__ out) {
  int m = blockIdx.x * 256 + threadIdx.x;
  if (m >= MROWS) return;
  int b = m / NPTS, n = m % NPTS;
  float x0 = X[((size_t)b * 3 + 0) * NPTS + n];
  float x1 = X[((size_t)b * 3 + 1) * NPTS + n];
  float x2 = X[((size_t)b * 3 + 2) * NPTS + n];
  const float* Tb = &T[b * 9];
#pragma unroll
  for (int f = 0; f < 3; ++f)
    out[(size_t)m * 3 + f] = x0 * Tb[f] + x1 * Tb[3 + f] + x2 * Tb[6 + f];
}

__global__ void tnet128_apply(const float* __restrict__ inter, const float* __restrict__ Ft,
                              float* __restrict__ out) {
  int idx = blockIdx.x * 256 + threadIdx.x;
  if (idx >= MROWS * 128) return;
  int m = idx >> 7, f = idx & 127;
  int b = m / NPTS;
  const float* row = &inter[(size_t)m * 128];
  const float* Fb  = &Ft[(size_t)b * 16384];
  float s = 0.0f;
  for (int c = 0; c < 128; ++c) s += row[c] * Fb[c * 128 + f];
  out[idx] = s;
}

// ---------------------------------------------------------------------------
// Concat [l1|l2|l3|l4|l5|gf|cl] -> [M, 2960]
// ---------------------------------------------------------------------------
__global__ void concat_kernel(const float* __restrict__ l1, const float* __restrict__ l2,
                              const float* __restrict__ l3, const float* __restrict__ l4,
                              const float* __restrict__ l5, const float* __restrict__ gf,
                              const float* __restrict__ cl, float* __restrict__ out) {
  int idx = blockIdx.x * 256 + threadIdx.x;
  if (idx >= MROWS * 2960) return;
  int m = idx / 2960, c = idx % 2960;
  int b = m / NPTS;
  float v;
  if      (c <   64) v = l1[(size_t)m * 64   + c];
  else if (c <  128) v = l2[(size_t)m * 64   + (c - 64)];
  else if (c <  256) v = l3[(size_t)m * 128  + (c - 128)];
  else if (c <  384) v = l4[(size_t)m * 128  + (c - 256)];
  else if (c <  896) v = l5[(size_t)m * 512  + (c - 384)];
  else if (c < 2944) v = gf[(size_t)b * 2048 + (c - 896)];
  else               v = cl[(size_t)b * NUMOBJ + (c - 2944)];
  out[idx] = v;
}

// Final transpose: [M,50] -> (B,50,N)
__global__ void writeout(const float* __restrict__ Y, float* __restrict__ out) {
  int idx = blockIdx.x * 256 + threadIdx.x;
  if (idx >= MROWS * 50) return;
  int m = idx / 50, c = idx % 50;
  int b = m / NPTS, n = m % NPTS;
  out[((size_t)b * 50 + c) * NPTS + n] = Y[idx];
}

// ---------------------------------------------------------------------------
// Host orchestration
// ---------------------------------------------------------------------------
struct KS { int w; int I; int O; };

extern "C" void kernel_launch(void* const* d_in, const int* in_sizes, int n_in,
                              void* d_out, int out_size, void* d_ws, size_t ws_size,
                              hipStream_t stream) {
  (void)in_sizes; (void)n_in; (void)out_size; (void)ws_size;

  const float* x  = (const float*)d_in[0];
  const float* cl = (const float*)d_in[1];
  const float* W[21];
  for (int i = 0; i < 21; ++i) W[i] = (const float*)d_in[2 + i];
  // BN params: (gamma, beta) pairs starting at d_in[23], in BNS dict order
  auto bng = [&](int k) { return (const float*)d_in[23 + 2 * k]; };
  auto bnb = [&](int k) { return (const float*)d_in[23 + 2 * k + 1]; };

  // kshared layers (weight index, I, O) in execution order
  static const KS ks[15] = {
    {0, 3, 64}, {1, 64, 128}, {2, 128, 1024},          // IT1..IT3
    {6, 3, 64}, {7, 64, 64}, {8, 64, 128},             // j1..j3
    {9, 128, 128}, {10, 128, 128}, {11, 128, 1024},    // FT1..FT3
    {15, 128, 512}, {16, 512, 2048}, {17, 2960, 512},  // j4, j5, j6
    {18, 512, 256}, {19, 256, 128}, {20, 128, 50}      // j7, j8, j9
  };

  // Workspace bump allocator
  size_t off = 0;
  auto alloc  = [&](size_t bytes) -> void* {
    void* p = (char*)d_ws + off;
    off += (bytes + 255) & ~(size_t)255;
    return p;
  };
  auto allocf = [&](size_t elems) { return (float*)alloc(elems * sizeof(float)); };

  // bf16 GEMM-ready transposed weights [Opad, Kpad]
  bf16* Wb[15];
  for (int li = 0; li < 15; ++li) {
    int Kpad = KP(ks[li].I), Opad = OPD(ks[li].O);
    Wb[li] = (bf16*)alloc((size_t)Kpad * Opad * sizeof(bf16));
    int n = Kpad * Opad;
    wconv<<<cdiv(n, 256), 256, 0, stream>>>(W[ks[li].w], Wb[li], ks[li].I, ks[li].O, Kpad, Opad);
  }

  // Activations / scratch
  bf16*  Abuf  = (bf16*)alloc((size_t)MROWS * KP(2960) * sizeof(bf16)); // largest A matrix
  float* psum  = allocf(32 * 2048);
  float* psq   = allocf(32 * 2048);
  float* meanv = allocf(2048);
  float* rsigv = allocf(2048);
  float* jbuf  = allocf((size_t)BSZ * 1024 * 5);

  float* it1   = allocf((size_t)MROWS * 64);
  float* it2   = allocf((size_t)MROWS * 128);
  float* it3   = allocf((size_t)MROWS * 1024);
  float* feat1 = allocf((size_t)BSZ * 1024);
  float* kt1   = allocf((size_t)BSZ * 512);
  float* kt2   = allocf((size_t)BSZ * 256);
  float* tmat  = allocf((size_t)BSZ * 9);
  float* xt    = allocf((size_t)MROWS * 3);
  float* l1    = allocf((size_t)MROWS * 64);
  float* l2    = allocf((size_t)MROWS * 64);
  float* l3    = allocf((size_t)MROWS * 128);
  float* f1    = allocf((size_t)MROWS * 128);
  float* f2    = allocf((size_t)MROWS * 128);
  float* f3    = allocf((size_t)MROWS * 1024);
  float* feat2 = allocf((size_t)BSZ * 1024);
  float* kf1   = allocf((size_t)BSZ * 512);
  float* kf2   = allocf((size_t)BSZ * 256);
  float* ftm   = allocf((size_t)BSZ * 16384);
  float* l4    = allocf((size_t)MROWS * 128);
  float* l5    = allocf((size_t)MROWS * 512);
  float* y5    = allocf((size_t)MROWS * 2048);
  float* gf    = allocf((size_t)BSZ * 2048);
  float* hcat  = allocf((size_t)MROWS * 2960);
  float* y6    = allocf((size_t)MROWS * 512);
  float* y7    = allocf((size_t)MROWS * 256);
  float* y8    = allocf((size_t)MROWS * 128);
  float* y9    = allocf((size_t)MROWS * 50);

  auto run_ks = [&](const float* in, int bcn, int li, float* Y, int bnk) {
    int I = ks[li].I, O = ks[li].O;
    int Kpad = KP(I), Opad = OPD(O);
    int ne = MROWS * I;
    if (bcn) expand_bcn<<<cdiv(ne, 256), 256, 0, stream>>>(in, Abuf, I, Kpad);
    else     expand_mc <<<cdiv(ne, 256), 256, 0, stream>>>(in, Abuf, I, Kpad);
    int k5 = 5 * I;
    if (Kpad > k5) {
      int np = MROWS * (Kpad - k5);
      padk<<<cdiv(np, 256), 256, 0, stream>>>(Abuf, k5, Kpad);
    }
    dim3 g(Opad / BNN, MROWS / BM);
    gemm_bf16_wmma<<<g, 256, 0, stream>>>(Abuf, Wb[li], Y, Kpad, O, Opad);
    if (bnk >= 0) {
      bn_stats1<<<cdiv(O * 32, 256), 256, 0, stream>>>(Y, O, psum, psq);
      bn_stats2<<<cdiv(O, 256), 256, 0, stream>>>(psum, psq, O, meanv, rsigv);
      bn_apply<<<cdiv(MROWS * O, 256), 256, 0, stream>>>(Y, O, meanv, rsigv, bng(bnk), bnb(bnk));
    }
  };
  auto run_kan = [&](const float* in, int wi, int I, int O, float* Y, int bnk) {
    kan_expand<<<cdiv(BSZ * I, 256), 256, 0, stream>>>(in, jbuf, I);
    kan_gemv<<<cdiv(BSZ * O, 256), 256, 0, stream>>>(jbuf, W[wi], Y, I, O);
    if (bnk >= 0)
      bn_batch<<<cdiv(O, 256), 256, 0, stream>>>(Y, O, bng(bnk), bnb(bnk));
  };

  // ---- input T-Net (3x3) ----
  run_ks(x,   1, 0, it1, 0);            // IT1 + ITbn1
  run_ks(it1, 0, 1, it2, 1);            // IT2 + ITbn2
  run_ks(it2, 0, 2, it3, 2);            // IT3 + ITbn3
  maxpool<<<cdiv(BSZ * 1024, 256), 256, 0, stream>>>(it3, 1024, feat1);
  run_kan(feat1, 3, 1024, 512, kt1, 3); // IT4 + ITbn4
  run_kan(kt1,   4, 512,  256, kt2, 4); // IT5 + ITbn5
  run_kan(kt2,   5, 256,  9,   tmat, 5);// IT6 + ITbn6 -> T (B,3,3)
  tnet3_apply<<<cdiv(MROWS, 256), 256, 0, stream>>>(x, tmat, xt);

  // ---- shared MLP 1 ----
  run_ks(xt, 0, 3, l1, 6);              // j1 + bn1
  run_ks(l1, 0, 4, l2, 7);              // j2 + bn2
  run_ks(l2, 0, 5, l3, 8);              // j3 + bn3  (= inter)

  // ---- feature T-Net (128x128) ----
  run_ks(l3, 0, 6, f1, 9);              // FT1 + FTbn1
  run_ks(f1, 0, 7, f2, 10);             // FT2 + FTbn2
  run_ks(f2, 0, 8, f3, 11);             // FT3 + FTbn3
  maxpool<<<cdiv(BSZ * 1024, 256), 256, 0, stream>>>(f3, 1024, feat2);
  run_kan(feat2, 12, 1024, 512,   kf1, 12); // FT4 + FTbn4
  run_kan(kf1,   13, 512,  256,   kf2, 13); // FT5 + FTbn5
  run_kan(kf2,   14, 256,  16384, ftm, 14); // FT6 + FTbn6 -> Ft (B,128,128)
  tnet128_apply<<<cdiv(MROWS * 128, 256), 256, 0, stream>>>(l3, ftm, l4);

  // ---- shared MLP 2 + global feature ----
  run_ks(l4, 0, 9,  l5, 15);            // j4 + bn4
  run_ks(l5, 0, 10, y5, 16);            // j5 + bn5
  maxpool<<<cdiv(BSZ * 2048, 256), 256, 0, stream>>>(y5, 2048, gf);
  concat_kernel<<<cdiv(MROWS * 2960, 256), 256, 0, stream>>>(l1, l2, l3, l4, l5, gf, cl, hcat);

  // ---- segmentation head ----
  run_ks(hcat, 0, 11, y6, 17);          // j6 + bn6
  run_ks(y6,   0, 12, y7, 18);          // j7 + bn7
  run_ks(y7,   0, 13, y8, 19);          // j8 + bn8
  run_ks(y8,   0, 14, y9, -1);          // j9 (no BN)
  writeout<<<cdiv(MROWS * 50, 256), 256, 0, stream>>>(y9, (float*)d_out);
}
